// SuperGlue_43284680409338
// MI455X (gfx1250) — matrix operand order repaired and checked by reading, not compiled, not run on hardware
//
#include <hip/hip_runtime.h>
#include <hip/hip_bf16.h>

// ---------------------------------------------------------------------------
// SuperGlue forward for MI455X (gfx1250, wave32, WMMA).
// f16 WMMA (v_wmma_f32_16x16x32_f16) with f32 accumulate; double-buffered LDS
// tiles; float4 (b128) global fills; fused epilogues; layer-invariant rank
// matrix precomputed once for the distance-modulated attention.
// ---------------------------------------------------------------------------

typedef __attribute__((ext_vector_type(16))) _Float16 v16h;
typedef __attribute__((ext_vector_type(8)))  _Float16 v8h;
typedef __attribute__((ext_vector_type(4)))  _Float16 v4h;
typedef __attribute__((ext_vector_type(8)))  float    v8f;

constexpr int BB   = 4;     // batch
constexpr int NPT  = 512;   // keypoints per image
constexpr int DD   = 256;   // descriptor dim
constexpr int NHD  = 4;     // heads
constexpr int NLAY = 18;    // gnn layers
constexpr int NP1  = 513;   // N+1 (sinkhorn)
constexpr int SINK = 100;

constexpr int LDA = 40;     // padded f16 row stride, A tile [64 rows][32 k]
constexpr int LDB = 40;     // padded f16 col stride, B tile stored [n][k]
constexpr long DN = (long)DD * NPT;  // per-batch activation stride (floats)

// ---- WMMA fragment loads (CDNA5 documented lane layouts) -------------------
// A (16x32 f16): lane L: M=L&15, half=L>>4; regs 0..3 hold K=half*8+{0..7},
// regs 4..7 hold K=16+half*8+{0..7}.  Two 16B ds_load_b128 per fragment.
__device__ inline v16h frag_a(const _Float16* As, int lane) {
  int m = lane & 15, lh = lane >> 4;
  const _Float16* row = As + m * LDA;
  v8h lo = *(const v8h*)(row + lh * 8);
  v8h hi = *(const v8h*)(row + 16 + lh * 8);
  v16h f;
#pragma unroll
  for (int e = 0; e < 8; ++e) { f[e] = lo[e]; f[8 + e] = hi[e]; }
  return f;
}
// B (32x16 f16): lane L: N=L&15, half=L>>4; elements 0..7 hold K=half*8+{0..7},
// elements 8..15 hold K=16+half*8+{0..7} (same order as A).
__device__ inline v16h frag_b(const _Float16* Bs, int lane) {
  int n = lane & 15, lh = lane >> 4;
  const _Float16* col = Bs + n * LDB;
  v8h lo = *(const v8h*)(col + lh * 8);
  v8h hi = *(const v8h*)(col + 16 + lh * 8);
  v16h f;
#pragma unroll
  for (int e = 0; e < 8; ++e) { f[e] = lo[e]; f[8 + e] = hi[e]; }
  return f;
}
__device__ inline v8f wmma_f16(v16h a, v16h b, v8f c) {
  return __builtin_amdgcn_wmma_f32_16x16x32_f16(false, a, false, b,
                                                (short)0, c, false, false);
}
__device__ inline v4h cvt4(float4 f) {
  v4h h;
  h[0] = (_Float16)f.x; h[1] = (_Float16)f.y;
  h[2] = (_Float16)f.z; h[3] = (_Float16)f.w;
  return h;
}

// ===========================================================================
// Generic GEMM: Y[b] = W(MxK) * X[b](KxN=512) + bias (+ residual)
// Split-K X source (fused concat): rows [0,ksplit) from X1, rest from X2.
// ksplit is always a multiple of 32 => source select is uniform per k-tile.
// Block 256 thr = 8 waves, tile 64x64, BK=32, double-buffered LDS.
// ===========================================================================
struct GemmP {
  const float* W; const float* bias;
  const float* X1; const float* X2; int ksplit;
  const float* resid;
  float* Y;
  int M, K;
  long xbs1, xbs2, ybs, rbs;   // per-batch strides (floats)
};

__global__ __launch_bounds__(256) void gemm_wmma(GemmP p) {
  __shared__ _Float16 As[2][64 * LDA];
  __shared__ _Float16 Bs[2][64 * LDB];
  const int t = threadIdx.x, lane = t & 31, wave = t >> 5;
  const int wm = wave >> 1, wn = wave & 1;
  const int b  = blockIdx.z;
  const int n0 = blockIdx.x * 64, m0 = blockIdx.y * 64;
  const float* X1 = p.X1 + (long)b * p.xbs1;
  const float* X2 = p.X2 + (long)b * p.xbs2;
  float*       Y  = p.Y  + (long)b * p.ybs;

  // fill coords: A 64x32 (2 float4/thread, contiguous k), B 32x64 (contig n)
  const int ar0 = t >> 3,         ac0 = (t & 7) << 2;
  const int ar1 = (t + 256) >> 3, ac1 = ((t + 256) & 7) << 2;
  const int bk0 = t >> 4,         bc0 = (t & 15) << 2;
  const int bk1 = (t + 256) >> 4, bc1 = ((t + 256) & 15) << 2;

  float4 aw0, aw1, bw0, bw1;
  auto loadT = [&](int kt) {
    const int k0 = kt << 5;
    aw0 = *(const float4*)&p.W[(long)(m0 + ar0) * p.K + k0 + ac0];
    aw1 = *(const float4*)&p.W[(long)(m0 + ar1) * p.K + k0 + ac1];
    const float* base = (k0 < p.ksplit) ? (X1 + (long)k0 * NPT)
                                        : (X2 + (long)(k0 - p.ksplit) * NPT);
    bw0 = *(const float4*)&base[(long)bk0 * NPT + n0 + bc0];
    bw1 = *(const float4*)&base[(long)bk1 * NPT + n0 + bc1];
  };
  auto storeT = [&](int buf) {
    _Float16* A  = As[buf];
    _Float16* Bm = Bs[buf];
    *(v4h*)&A[ar0 * LDA + ac0] = cvt4(aw0);      // ds_store_b64
    *(v4h*)&A[ar1 * LDA + ac1] = cvt4(aw1);
    Bm[(bc0 + 0) * LDB + bk0] = (_Float16)bw0.x; // transposed scatter
    Bm[(bc0 + 1) * LDB + bk0] = (_Float16)bw0.y;
    Bm[(bc0 + 2) * LDB + bk0] = (_Float16)bw0.z;
    Bm[(bc0 + 3) * LDB + bk0] = (_Float16)bw0.w;
    Bm[(bc1 + 0) * LDB + bk1] = (_Float16)bw1.x;
    Bm[(bc1 + 1) * LDB + bk1] = (_Float16)bw1.y;
    Bm[(bc1 + 2) * LDB + bk1] = (_Float16)bw1.z;
    Bm[(bc1 + 3) * LDB + bk1] = (_Float16)bw1.w;
  };

  const int ksteps = p.K >> 5;
  loadT(0); storeT(0);
  int cur = 0;
  v8f acc0 = {}, acc1 = {};
  for (int kt = 0; kt < ksteps; ++kt) {
    __syncthreads();
    const bool more = (kt + 1 < ksteps);
    if (more) loadT(kt + 1);                       // overlap with wmma
    if (kt + 2 < ksteps)                           // global_prefetch_b8
      __builtin_prefetch(&p.W[(long)m0 * p.K + ((kt + 2) << 5)], 0, 1);
    v16h a  = frag_a(&As[cur][wm * 16 * LDA], lane);
    v16h b0 = frag_b(&Bs[cur][(wn * 32) * LDB], lane);
    acc0 = wmma_f16(a, b0, acc0);
    v16h b1 = frag_b(&Bs[cur][(wn * 32 + 16) * LDB], lane);
    acc1 = wmma_f16(a, b1, acc1);
    if (more) storeT(cur ^ 1);
    cur ^= 1;
  }
  const int lh = lane >> 4, nl = lane & 15;
#pragma unroll
  for (int r = 0; r < 8; ++r) {
    int gm = m0 + wm * 16 + r + 8 * lh;            // C layout: VGPR r, half
    float bv = p.bias ? p.bias[gm] : 0.f;
#pragma unroll
    for (int tt = 0; tt < 2; ++tt) {
      int gn = n0 + wn * 32 + tt * 16 + nl;
      float v = (tt ? acc1[r] : acc0[r]) + bv;
      if (p.resid) v += p.resid[(long)b * p.rbs + (long)gm * NPT + gn];
      Y[(long)gm * NPT + gn] = v;
    }
  }
}

// ===========================================================================
// Attention scores: S[b,h,n,m] = (q_n . k_m) * pd[n, rank[n,m]] * dist[n,m]/8
// A = Q^T (n x dh), B = K (dh x m), K=64. Heads interleaved: d = dh*NH + h.
// ===========================================================================
struct ScoreP {
  const float* q; const float* k;                 // [B][D][N]
  const float* dist; const unsigned short* rank;  // [B][N][N] (combo slice)
  const float* pd;                                // [N][N]
  float* S;                                       // [B][NH][N][N]
};
__global__ __launch_bounds__(256) void attn_score(ScoreP p) {
  __shared__ _Float16 As[2][64 * LDA];
  __shared__ _Float16 Bs[2][64 * LDB];
  const int t = threadIdx.x, lane = t & 31, wave = t >> 5;
  const int wm = wave >> 1, wn = wave & 1;
  const int bh = blockIdx.z, b = bh >> 2, h = bh & 3;
  const int m0 = blockIdx.x * 64, n0 = blockIdx.y * 64;
  const float* qb = p.q + (long)b * DN;
  const float* kb = p.k + (long)b * DN;

  // A[n][dh]=Q^T: contiguous along n; B[dh][m]: contiguous along m
  const int ac0 = t >> 4,         ar0 = (t & 15) << 2;
  const int ac1 = (t + 256) >> 4, ar1 = ((t + 256) & 15) << 2;
  const int bk0 = t >> 4,         bc0 = (t & 15) << 2;
  const int bk1 = (t + 256) >> 4, bc1 = ((t + 256) & 15) << 2;

  float4 aw0, aw1, bw0, bw1;
  auto loadT = [&](int kt) {
    const int k0 = kt << 5;
    aw0 = *(const float4*)&qb[((long)(k0 + ac0) * NHD + h) * NPT + n0 + ar0];
    aw1 = *(const float4*)&qb[((long)(k0 + ac1) * NHD + h) * NPT + n0 + ar1];
    bw0 = *(const float4*)&kb[((long)(k0 + bk0) * NHD + h) * NPT + m0 + bc0];
    bw1 = *(const float4*)&kb[((long)(k0 + bk1) * NHD + h) * NPT + m0 + bc1];
  };
  auto storeT = [&](int buf) {
    _Float16* A  = As[buf];
    _Float16* Bm = Bs[buf];
    A[(ar0 + 0) * LDA + ac0] = (_Float16)aw0.x;
    A[(ar0 + 1) * LDA + ac0] = (_Float16)aw0.y;
    A[(ar0 + 2) * LDA + ac0] = (_Float16)aw0.z;
    A[(ar0 + 3) * LDA + ac0] = (_Float16)aw0.w;
    A[(ar1 + 0) * LDA + ac1] = (_Float16)aw1.x;
    A[(ar1 + 1) * LDA + ac1] = (_Float16)aw1.y;
    A[(ar1 + 2) * LDA + ac1] = (_Float16)aw1.z;
    A[(ar1 + 3) * LDA + ac1] = (_Float16)aw1.w;
    Bm[(bc0 + 0) * LDB + bk0] = (_Float16)bw0.x;
    Bm[(bc0 + 1) * LDB + bk0] = (_Float16)bw0.y;
    Bm[(bc0 + 2) * LDB + bk0] = (_Float16)bw0.z;
    Bm[(bc0 + 3) * LDB + bk0] = (_Float16)bw0.w;
    Bm[(bc1 + 0) * LDB + bk1] = (_Float16)bw1.x;
    Bm[(bc1 + 1) * LDB + bk1] = (_Float16)bw1.y;
    Bm[(bc1 + 2) * LDB + bk1] = (_Float16)bw1.z;
    Bm[(bc1 + 3) * LDB + bk1] = (_Float16)bw1.w;
  };

  loadT(0); storeT(0);
  int cur = 0;
  v8f acc0 = {}, acc1 = {};
#pragma unroll
  for (int kt = 0; kt < 2; ++kt) {
    __syncthreads();
    if (kt == 0) loadT(1);
    v16h a  = frag_a(&As[cur][wm * 16 * LDA], lane);
    v16h b0 = frag_b(&Bs[cur][(wn * 32) * LDB], lane);
    acc0 = wmma_f16(a, b0, acc0);
    v16h b1 = frag_b(&Bs[cur][(wn * 32 + 16) * LDB], lane);
    acc1 = wmma_f16(a, b1, acc1);
    if (kt == 0) storeT(1);
    cur ^= 1;
  }
  const int lh = lane >> 4, nl = lane & 15;
  float* Sb = p.S + (long)(b * NHD + h) * NPT * NPT;
  const float* db = p.dist + (long)b * NPT * NPT;
  const unsigned short* rb = p.rank + (long)b * NPT * NPT;
#pragma unroll
  for (int r = 0; r < 8; ++r) {
    int gn = n0 + wm * 16 + r + 8 * lh;
#pragma unroll
    for (int tt = 0; tt < 2; ++tt) {
      int gm = m0 + wn * 32 + tt * 16 + nl;
      long e = (long)gn * NPT + gm;
      float dmod = db[e] * p.pd[(long)gn * NPT + rb[e]];
      Sb[e] = (tt ? acc1[r] : acc0[r]) * dmod * 0.125f;   // 1/sqrt(64)
    }
  }
}

// ===========================================================================
// Message: msg[b, dh*NH+h, n] = sum_m V[dh*NH+h, m] * P[b,h,n,m]
// A = V (64 x 512), B = P^T (m x n), K=512. Fully b128-vectorized fills.
// ===========================================================================
struct MsgP { const float* v; const float* P; float* msg; };
__global__ __launch_bounds__(256) void msg_gemm(MsgP p) {
  __shared__ _Float16 As[2][64 * LDA];
  __shared__ _Float16 Bs[2][64 * LDB];
  const int t = threadIdx.x, lane = t & 31, wave = t >> 5;
  const int wm = wave >> 1, wn = wave & 1;
  const int bh = blockIdx.z, b = bh >> 2, h = bh & 3;
  const int n0 = blockIdx.x * 64;
  const float* vb = p.v + (long)b * DN;
  const float* Pb = p.P + (long)(b * NHD + h) * NPT * NPT;

  // A[dh][m]: contiguous along m(k); B[m][n]=P[n][m]: contiguous along m(k)
  const int ar0 = t >> 3,         ac0 = (t & 7) << 2;
  const int ar1 = (t + 256) >> 3, ac1 = ((t + 256) & 7) << 2;
  const int bc0 = t >> 3,         bk0 = (t & 7) << 2;
  const int bc1 = (t + 256) >> 3, bk1 = ((t + 256) & 7) << 2;

  float4 aw0, aw1, bw0, bw1;
  auto loadT = [&](int kt) {
    const int k0 = kt << 5;
    aw0 = *(const float4*)&vb[((long)ar0 * NHD + h) * NPT + k0 + ac0];
    aw1 = *(const float4*)&vb[((long)ar1 * NHD + h) * NPT + k0 + ac1];
    bw0 = *(const float4*)&Pb[(long)(n0 + bc0) * NPT + k0 + bk0];
    bw1 = *(const float4*)&Pb[(long)(n0 + bc1) * NPT + k0 + bk1];
  };
  auto storeT = [&](int buf) {
    _Float16* A  = As[buf];
    _Float16* Bm = Bs[buf];
    *(v4h*)&A[ar0 * LDA + ac0]  = cvt4(aw0);   // ds_store_b64
    *(v4h*)&A[ar1 * LDA + ac1]  = cvt4(aw1);
    *(v4h*)&Bm[bc0 * LDB + bk0] = cvt4(bw0);
    *(v4h*)&Bm[bc1 * LDB + bk1] = cvt4(bw1);
  };

  loadT(0); storeT(0);
  int cur = 0;
  v8f acc0 = {}, acc1 = {};
  for (int kt = 0; kt < 16; ++kt) {
    __syncthreads();
    const bool more = (kt + 1 < 16);
    if (more) loadT(kt + 1);
    v16h a  = frag_a(&As[cur][wm * 16 * LDA], lane);
    v16h b0 = frag_b(&Bs[cur][(wn * 32) * LDB], lane);
    acc0 = wmma_f16(a, b0, acc0);
    v16h b1 = frag_b(&Bs[cur][(wn * 32 + 16) * LDB], lane);
    acc1 = wmma_f16(a, b1, acc1);
    if (more) storeT(cur ^ 1);
    cur ^= 1;
  }
  const int lh = lane >> 4, nl = lane & 15;
  float* mb = p.msg + (long)b * DN;
#pragma unroll
  for (int r = 0; r < 8; ++r) {
    int dh = wm * 16 + r + 8 * lh;
#pragma unroll
    for (int tt = 0; tt < 2; ++tt) {
      int gn = n0 + wn * 32 + tt * 16 + nl;
      mb[((long)dh * NHD + h) * NPT + gn] = tt ? acc1[r] : acc0[r];
    }
  }
}

// ===========================================================================
// Final matching scores -> couplings[b, n, m] = (m0^T m1)/16, row stride 513
// ===========================================================================
struct FinP { const float* m0; const float* m1; float* C; };
__global__ __launch_bounds__(256) void final_score(FinP p) {
  __shared__ _Float16 As[2][64 * LDA];
  __shared__ _Float16 Bs[2][64 * LDB];
  const int t = threadIdx.x, lane = t & 31, wave = t >> 5;
  const int wm = wave >> 1, wn = wave & 1;
  const int b = blockIdx.z;
  const int m0c = blockIdx.x * 64, n0 = blockIdx.y * 64;
  const float* Am = p.m0 + (long)b * DN;
  const float* Bg = p.m1 + (long)b * DN;

  const int ac0 = t >> 4,         ar0 = (t & 15) << 2;   // A: contig n
  const int ac1 = (t + 256) >> 4, ar1 = ((t + 256) & 15) << 2;
  const int bk0 = t >> 4,         bc0 = (t & 15) << 2;   // B: contig m
  const int bk1 = (t + 256) >> 4, bc1 = ((t + 256) & 15) << 2;

  float4 aw0, aw1, bw0, bw1;
  auto loadT = [&](int kt) {
    const int k0 = kt << 5;
    aw0 = *(const float4*)&Am[(long)(k0 + ac0) * NPT + n0 + ar0];
    aw1 = *(const float4*)&Am[(long)(k0 + ac1) * NPT + n0 + ar1];
    bw0 = *(const float4*)&Bg[(long)(k0 + bk0) * NPT + m0c + bc0];
    bw1 = *(const float4*)&Bg[(long)(k0 + bk1) * NPT + m0c + bc1];
  };
  auto storeT = [&](int buf) {
    _Float16* A  = As[buf];
    _Float16* Bm = Bs[buf];
    A[(ar0 + 0) * LDA + ac0] = (_Float16)aw0.x;
    A[(ar0 + 1) * LDA + ac0] = (_Float16)aw0.y;
    A[(ar0 + 2) * LDA + ac0] = (_Float16)aw0.z;
    A[(ar0 + 3) * LDA + ac0] = (_Float16)aw0.w;
    A[(ar1 + 0) * LDA + ac1] = (_Float16)aw1.x;
    A[(ar1 + 1) * LDA + ac1] = (_Float16)aw1.y;
    A[(ar1 + 2) * LDA + ac1] = (_Float16)aw1.z;
    A[(ar1 + 3) * LDA + ac1] = (_Float16)aw1.w;
    Bm[(bc0 + 0) * LDB + bk0] = (_Float16)bw0.x;
    Bm[(bc0 + 1) * LDB + bk0] = (_Float16)bw0.y;
    Bm[(bc0 + 2) * LDB + bk0] = (_Float16)bw0.z;
    Bm[(bc0 + 3) * LDB + bk0] = (_Float16)bw0.w;
    Bm[(bc1 + 0) * LDB + bk1] = (_Float16)bw1.x;
    Bm[(bc1 + 1) * LDB + bk1] = (_Float16)bw1.y;
    Bm[(bc1 + 2) * LDB + bk1] = (_Float16)bw1.z;
    Bm[(bc1 + 3) * LDB + bk1] = (_Float16)bw1.w;
  };

  loadT(0); storeT(0);
  int cur = 0;
  v8f acc0 = {}, acc1 = {};
  for (int kt = 0; kt < 8; ++kt) {
    __syncthreads();
    const bool more = (kt + 1 < 8);
    if (more) loadT(kt + 1);
    v16h a  = frag_a(&As[cur][wm * 16 * LDA], lane);
    v16h b0 = frag_b(&Bs[cur][(wn * 32) * LDB], lane);
    acc0 = wmma_f16(a, b0, acc0);
    v16h b1 = frag_b(&Bs[cur][(wn * 32 + 16) * LDB], lane);
    acc1 = wmma_f16(a, b1, acc1);
    if (more) storeT(cur ^ 1);
    cur ^= 1;
  }
  const int lh = lane >> 4, nl = lane & 15;
  float* Cb = p.C + (long)b * NP1 * NP1;
#pragma unroll
  for (int r = 0; r < 8; ++r) {
    int gn = n0 + wm * 16 + r + 8 * lh;
#pragma unroll
    for (int tt = 0; tt < 2; ++tt) {
      int gm = m0c + wn * 32 + tt * 16 + nl;
      Cb[(long)gn * NP1 + gm] = (tt ? acc1[r] : acc0[r]) * 0.0625f; // 1/sqrt(256)
    }
  }
}

// ===========================================================================
// Keypoint encoder: 5 tiny 1x1 convs + LN(ddof=1)+relu, fused; adds desc.
// ===========================================================================
struct KencP {
  const float* w[5]; const float* bs[5]; const float* la[4]; const float* lb[4];
  const float* desc; const float* kp; const float* sc;
  float* dout; float* kn;
};
__global__ __launch_bounds__(256) void kenc_kernel(KencP p) {
  __shared__ float A0[256 * 17];
  __shared__ float A1[256 * 17];
  __shared__ float rs[16][16], rq[16][16], ms[2][16];
  const int CH[6] = {3, 32, 64, 128, 256, 256};
  const int t = threadIdx.x, b = blockIdx.y, n0 = blockIdx.x * 16;
  const int p16 = t & 15, part = t >> 4;
  if (t < 16) {
    int n = n0 + t;
    float kx = (p.kp[((long)b * NPT + n) * 2 + 0] - 320.f) * (1.f / 448.f);
    float ky = (p.kp[((long)b * NPT + n) * 2 + 1] - 240.f) * (1.f / 448.f);
    p.kn[((long)b * NPT + n) * 2 + 0] = kx;
    p.kn[((long)b * NPT + n) * 2 + 1] = ky;
    A0[0 * 17 + t] = kx; A0[1 * 17 + t] = ky;
    A0[2 * 17 + t] = p.sc[(long)b * NPT + n];
  }
  __syncthreads();
  float* src = A0; float* dst = A1;
  for (int L = 0; L < 5; ++L) {
    const int Ci = CH[L], Co = CH[L + 1];
    const float* W = p.w[L]; const float* bias = p.bs[L];
    for (int idx = t; idx < Co * 16; idx += 256) {
      int o = idx >> 4, pp = idx & 15;
      float s = bias[o];
      for (int c = 0; c < Ci; ++c) s += W[o * Ci + c] * src[c * 17 + pp];
      if (L < 4) dst[o * 17 + pp] = s;
      else {
        long gi = ((long)b * DD + o) * NPT + (n0 + pp);
        p.dout[gi] = p.desc[gi] + s;
      }
    }
    __syncthreads();
    if (L < 4) {
      float s = 0.f, q = 0.f;
      for (int c = part; c < Co; c += 16) { float v = dst[c * 17 + p16]; s += v; q += v * v; }
      rs[part][p16] = s; rq[part][p16] = q;
      __syncthreads();
      if (t < 16) {
        float S = 0.f, Q = 0.f;
        for (int i = 0; i < 16; ++i) { S += rs[i][t]; Q += rq[i][t]; }
        float mean = S / Co;
        float var = (Q - Co * mean * mean) / (Co - 1);
        ms[0][t] = mean;
        ms[1][t] = sqrtf(fmaxf(var, 0.f)) + 1e-6f;
      }
      __syncthreads();
      const float* a = p.la[L]; const float* bb = p.lb[L];
      for (int idx = t; idx < Co * 16; idx += 256) {
        int o = idx >> 4, pp = idx & 15;
        float v = a[o] * ((dst[o * 17 + pp] - ms[0][pp]) / ms[1][pp]) + bb[o];
        dst[o * 17 + pp] = v > 0.f ? v : 0.f;
      }
      __syncthreads();
      float* tmp = src; src = dst; dst = tmp;
    }
  }
}

// ===========================================================================
// Distances + rank (argsort of argsort, stable ties) per (combo, b, row).
// ===========================================================================
struct DistP { const float* kq; const float* ks; float* dist; unsigned short* rank; };
__global__ __launch_bounds__(256) void dist_rank(DistP p) {
  __shared__ float sk[NPT * 2];
  __shared__ float dr[NPT];
  const int t = threadIdx.x, b = blockIdx.y, row = blockIdx.x;
  const float* ksb = p.ks + (long)b * NPT * 2;
  for (int i = t; i < NPT * 2; i += 256) sk[i] = ksb[i];
  __syncthreads();
  float qx = p.kq[((long)b * NPT + row) * 2 + 0];
  float qy = p.kq[((long)b * NPT + row) * 2 + 1];
  long base = ((long)b * NPT + row) * NPT;
  for (int m = t; m < NPT; m += 256) {
    float dx = qx - sk[m * 2], dy = qy - sk[m * 2 + 1];
    float sq = dx * dx + dy * dy;
    float d = sq > 0.f ? sqrtf(sq) : 0.f;
    dr[m] = d; p.dist[base + m] = d;
  }
  __syncthreads();
  for (int j = t; j < NPT; j += 256) {
    float dj = dr[j]; int cnt = 0;
    for (int i = 0; i < NPT; ++i) {
      float di = dr[i];
      cnt += (di < dj) || (di == dj && i < j);
    }
    p.rank[base + j] = (unsigned short)cnt;
  }
}

// ===========================================================================
// Row softmax over m (512) for S[b*NH + h][row][*]
// ===========================================================================
__global__ __launch_bounds__(256) void softmax_rows(float* S) {
  __shared__ float red[256];
  const int t = threadIdx.x;
  long base = ((long)blockIdx.y * NPT + blockIdx.x) * NPT;
  float x0 = S[base + t], x1 = S[base + t + 256];
  red[t] = fmaxf(x0, x1); __syncthreads();
  for (int s = 128; s > 0; s >>= 1) { if (t < s) red[t] = fmaxf(red[t], red[t + s]); __syncthreads(); }
  float mx = red[0]; __syncthreads();
  float e0 = __expf(x0 - mx), e1 = __expf(x1 - mx);
  red[t] = e0 + e1; __syncthreads();
  for (int s = 128; s > 0; s >>= 1) { if (t < s) red[t] += red[t + s]; __syncthreads(); }
  float inv = 1.f / red[0];
  S[base + t] = e0 * inv; S[base + t + 256] = e1 * inv;
}

// ===========================================================================
// LayerNorm over channel axis (C=512, unbiased std) + ReLU, in place.
// ===========================================================================
struct LnP { float* y; const float* a; const float* bt; int C; };
__global__ __launch_bounds__(256) void ln_relu(LnP p) {
  __shared__ float rs[8][32], rq[8][32], ms[2][32];
  const int t = threadIdx.x, part = t >> 5, pc = t & 31;
  const int pos = blockIdx.x * 32 + pc;
  long base = (long)blockIdx.y * p.C * NPT;
  float s = 0.f, q = 0.f;
  for (int c = part; c < p.C; c += 8) {
    float v = p.y[base + (long)c * NPT + pos]; s += v; q += v * v;
  }
  rs[part][pc] = s; rq[part][pc] = q; __syncthreads();
  if (t < 32) {
    float S = 0.f, Q = 0.f;
    for (int i = 0; i < 8; ++i) { S += rs[i][t]; Q += rq[i][t]; }
    float mean = S / p.C;
    float var = (Q - p.C * mean * mean) / (p.C - 1);
    ms[0][t] = mean; ms[1][t] = sqrtf(fmaxf(var, 0.f)) + 1e-6f;
  }
  __syncthreads();
  float mean = ms[0][pc], sd = ms[1][pc];
  for (int c = part; c < p.C; c += 8) {
    long i = base + (long)c * NPT + pos;
    float v = p.a[c] * ((p.y[i] - mean) / sd) + p.bt[c];
    p.y[i] = v > 0.f ? v : 0.f;
  }
}

// ===========================================================================
// Sinkhorn pieces
// ===========================================================================
__global__ __launch_bounds__(256) void fill_bins(float* C, const float* bin,
                                                 float* u, float* v) {
  int t = blockIdx.x * 256 + threadIdx.x;
  if (t >= BB * NP1) return;
  float bv = *bin;
  int b = t / NP1, i = t % NP1;
  float* Cb = C + (long)b * NP1 * NP1;
  Cb[(long)i * NP1 + (NP1 - 1)] = bv;
  Cb[(long)(NP1 - 1) * NP1 + i] = bv;
  u[t] = 0.f; v[t] = 0.f;
}

__global__ __launch_bounds__(256) void sink_u(const float* C, float* u, const float* v) {
  __shared__ float red[256];
  const int b = blockIdx.y, r = blockIdx.x, t = threadIdx.x;
  const float* Cb = C + (long)b * NP1 * NP1 + (long)r * NP1;
  const float* vb = v + (long)b * NP1;
  float mx = -3.0e38f;
  for (int m = t; m < NP1; m += 256) mx = fmaxf(mx, Cb[m] + vb[m]);
  red[t] = mx; __syncthreads();
  for (int s = 128; s > 0; s >>= 1) { if (t < s) red[t] = fmaxf(red[t], red[t + s]); __syncthreads(); }
  mx = red[0]; __syncthreads();
  float se = 0.f;
  for (int m = t; m < NP1; m += 256) se += __expf(Cb[m] + vb[m] - mx);
  red[t] = se; __syncthreads();
  for (int s = 128; s > 0; s >>= 1) { if (t < s) red[t] += red[t + s]; __syncthreads(); }
  if (t == 0) {
    const float norm = -logf(1024.f);
    float lmu = (r < NPT) ? norm : (logf(512.f) + norm);
    u[(long)b * NP1 + r] = lmu - (mx + logf(red[0]));
  }
}

__global__ __launch_bounds__(256) void sink_v(const float* C, float* v, const float* u) {
  __shared__ float red[256];
  const int b = blockIdx.y, c = blockIdx.x, t = threadIdx.x;
  const float* Cb = C + (long)b * NP1 * NP1 + c;
  const float* ub = u + (long)b * NP1;
  float mx = -3.0e38f;
  for (int n = t; n < NP1; n += 256) mx = fmaxf(mx, Cb[(long)n * NP1] + ub[n]);
  red[t] = mx; __syncthreads();
  for (int s = 128; s > 0; s >>= 1) { if (t < s) red[t] = fmaxf(red[t], red[t + s]); __syncthreads(); }
  mx = red[0]; __syncthreads();
  float se = 0.f;
  for (int n = t; n < NP1; n += 256) se += __expf(Cb[(long)n * NP1] + ub[n] - mx);
  red[t] = se; __syncthreads();
  for (int s = 128; s > 0; s >>= 1) { if (t < s) red[t] += red[t + s]; __syncthreads(); }
  if (t == 0) {
    const float norm = -logf(1024.f);
    float lnu = (c < NPT) ? norm : (logf(512.f) + norm);
    v[(long)b * NP1 + c] = lnu - (mx + logf(red[0]));
  }
}

__global__ __launch_bounds__(256) void final_out_k(const float* C, const float* u,
                                                   const float* v, float* out) {
  long i = (long)blockIdx.x * 256 + threadIdx.x;
  long tot = (long)BB * NP1 * NP1;
  if (i >= tot) return;
  long b = i / ((long)NP1 * NP1);
  long rem = i - b * NP1 * NP1;
  int r = (int)(rem / NP1), c = (int)(rem % NP1);
  const float norm = -logf(1024.f);
  out[i] = C[i] + u[b * NP1 + r] + v[b * NP1 + c] - norm;
}

// ===========================================================================
// Host orchestration (graph-capture safe: stream-ordered launches only).
// Input order: setup_inputs() dict order, nested containers flattened
// depth-first in insertion order:
//   0..5 desc0,desc1,kp0,kp1,sc0,sc1; 6..10 kenc ws; 11..15 kenc bs;
//   16..19 kenc ln_a; 20..23 kenc ln_b; 24+15*l+{0..14} per gnn layer
//   (wq,bq,wk,bk,wv,bv,wm,bm,pd,w1,b1,ln_a,ln_b,w2,b2); 294 final_w,
//   295 final_b, 296 bin_score, 297 height, 298 width (h/w hardcoded 480/640).
// ===========================================================================
extern "C" void kernel_launch(void* const* d_in, const int* in_sizes, int n_in,
                              void* d_out, int out_size, void* d_ws, size_t ws_size,
                              hipStream_t stream) {
  (void)in_sizes; (void)n_in; (void)out_size; (void)ws_size;
  auto in = [&](int i) { return (const float*)d_in[i]; };

  char* ws = (char*)d_ws;
  size_t off = 0;
  auto carve = [&](size_t bytes) -> char* {
    char* r = ws + off; off = (off + bytes + 255) & ~(size_t)255; return r;
  };
  const size_t ACT = (size_t)BB * DD * NPT * sizeof(float);      // 2 MB
  float* d0a   = (float*)carve(ACT);
  float* d1a   = (float*)carve(ACT);
  float* d0b   = (float*)carve(ACT);
  float* d1b   = (float*)carve(ACT);
  float* knorm = (float*)carve((size_t)2 * BB * NPT * 2 * sizeof(float));
  float* dists = (float*)carve((size_t)4 * BB * NPT * NPT * sizeof(float));
  unsigned short* rankb = (unsigned short*)carve((size_t)4 * BB * NPT * NPT * 2);
  float* qbuf  = (float*)carve(ACT);
  float* kbuf  = (float*)carve(ACT);
  float* vbuf  = (float*)carve(ACT);
  float* Sbuf  = (float*)carve((size_t)BB * NHD * NPT * NPT * sizeof(float));
  float* msgb  = (float*)carve(ACT);
  float* mproj = (float*)carve(ACT);
  float* y1    = (float*)carve((size_t)BB * 2 * DD * NPT * sizeof(float));
  float* m0f   = (float*)carve(ACT);
  float* m1f   = (float*)carve(ACT);
  float* coup  = (float*)carve((size_t)BB * NP1 * NP1 * sizeof(float));
  float* ub    = (float*)carve((size_t)BB * NP1 * sizeof(float));
  float* vb    = (float*)carve((size_t)BB * NP1 * sizeof(float));

  // --- keypoint encoder (both sides) ---------------------------------------
  for (int side = 0; side < 2; ++side) {
    KencP kp{};
    for (int i = 0; i < 5; ++i) { kp.w[i] = in(6 + i); kp.bs[i] = in(11 + i); }
    for (int i = 0; i < 4; ++i) { kp.la[i] = in(16 + i); kp.lb[i] = in(20 + i); }
    kp.desc = in(side ? 1 : 0);
    kp.kp   = in(side ? 3 : 2);
    kp.sc   = in(side ? 5 : 4);
    kp.dout = side ? d1a : d0a;
    kp.kn   = knorm + (size_t)side * BB * NPT * 2;
    kenc_kernel<<<dim3(NPT / 16, BB), 256, 0, stream>>>(kp);
  }

  // --- distances + rank, 4 combos: (0,0) (1,1) (0,1) (1,0) -----------------
  {
    const int qs[4] = {0, 1, 0, 1}, ss[4] = {0, 1, 1, 0};
    for (int cb = 0; cb < 4; ++cb) {
      DistP dp{knorm + (size_t)qs[cb] * BB * NPT * 2,
               knorm + (size_t)ss[cb] * BB * NPT * 2,
               dists + (long)cb * BB * NPT * NPT,
               rankb + (long)cb * BB * NPT * NPT};
      dist_rank<<<dim3(NPT, BB), 256, 0, stream>>>(dp);
    }
  }

  auto gemm = [&](const float* W, const float* bias, const float* X1,
                  const float* X2, int ksplit, const float* resid, float* Y,
                  int M, int K, long xbs1, long xbs2, long ybs, long rbs) {
    GemmP g{W, bias, X1, X2, ksplit, resid, Y, M, K, xbs1, xbs2, ybs, rbs};
    gemm_wmma<<<dim3(NPT / 64, M / 64, BB), 256, 0, stream>>>(g);
  };

  // --- GNN layers ----------------------------------------------------------
  float* cur[2] = {d0a, d1a};
  float* nxt[2] = {d0b, d1b};
  for (int l = 0; l < NLAY; ++l) {
    const int bi = 24 + 15 * l;
    const float *wq = in(bi + 0), *bq = in(bi + 1), *wk = in(bi + 2),
                *bk = in(bi + 3), *wv = in(bi + 4), *bvp = in(bi + 5),
                *wm = in(bi + 6), *bm = in(bi + 7), *pd = in(bi + 8),
                *w1 = in(bi + 9), *b1 = in(bi + 10), *lna = in(bi + 11),
                *lnb = in(bi + 12), *w2 = in(bi + 13), *b2 = in(bi + 14);
    const bool cross = (l & 1);
    for (int side = 0; side < 2; ++side) {
      const float* X   = cur[side];
      const float* SRC = cur[cross ? (1 - side) : side];
      gemm(wq, bq, X,   X,   DD, nullptr, qbuf, DD, DD, DN, DN, DN, 0);
      gemm(wk, bk, SRC, SRC, DD, nullptr, kbuf, DD, DD, DN, DN, DN, 0);
      gemm(wv, bvp, SRC, SRC, DD, nullptr, vbuf, DD, DD, DN, DN, DN, 0);
      const int combo = cross ? (2 + side) : side;
      ScoreP sp{qbuf, kbuf,
                dists + (long)combo * BB * NPT * NPT,
                rankb + (long)combo * BB * NPT * NPT, pd, Sbuf};
      attn_score<<<dim3(8, 8, BB * NHD), 256, 0, stream>>>(sp);
      softmax_rows<<<dim3(NPT, BB * NHD), 256, 0, stream>>>(Sbuf);
      MsgP mp{vbuf, Sbuf, msgb};
      msg_gemm<<<dim3(8, 1, BB * NHD), 256, 0, stream>>>(mp);
      gemm(wm, bm, msgb, msgb, DD, nullptr, mproj, DD, DD, DN, DN, DN, 0);
      // h = concat([x, msg_proj]) fused via split-K X sources
      gemm(w1, b1, X, mproj, DD, nullptr, y1, 2 * DD, 2 * DD, DN, DN, 2 * DN, 0);
      LnP lp{y1, lna, lnb, 2 * DD};
      ln_relu<<<dim3(NPT / 32, BB), 256, 0, stream>>>(lp);
      // d_new = w2*h + b2 + d_old
      gemm(w2, b2, y1, y1, 2 * DD, X, nxt[side], DD, 2 * DD, 2 * DN, 2 * DN, DN, DN);
    }
    float* t0 = cur[0]; cur[0] = nxt[0]; nxt[0] = t0;
    float* t1 = cur[1]; cur[1] = nxt[1]; nxt[1] = t1;
  }

  // --- final projection + couplings ---------------------------------------
  gemm(in(294), in(295), cur[0], cur[0], DD, nullptr, m0f, DD, DD, DN, DN, DN, 0);
  gemm(in(294), in(295), cur[1], cur[1], DD, nullptr, m1f, DD, DD, DN, DN, DN, 0);
  FinP fp{m0f, m1f, coup};
  final_score<<<dim3(8, 8, BB), 256, 0, stream>>>(fp);
  fill_bins<<<dim3((BB * NP1 + 255) / 256), 256, 0, stream>>>(coup, in(296), ub, vb);

  // --- Sinkhorn ------------------------------------------------------------
  for (int it = 0; it < SINK; ++it) {
    sink_u<<<dim3(NP1, BB), 256, 0, stream>>>(coup, ub, vb);
    sink_v<<<dim3(NP1, BB), 256, 0, stream>>>(coup, vb, ub);
  }
  long tot = (long)BB * NP1 * NP1;
  final_out_k<<<dim3((unsigned)((tot + 255) / 256)), 256, 0, stream>>>(
      coup, ub, vb, (float*)d_out);
}
